// MultiTaskGAT_8993661518653
// MI455X (gfx1250) — matrix operand order repaired and checked by reading, not compiled, not run on hardware
//
#include <hip/hip_runtime.h>
#include <hip/hip_bf16.h>
#include <stdint.h>

// Problem constants (from reference)
constexpr int kN  = 40000;          // nodes
constexpr int kE  = 160000;         // edges
constexpr int kD  = 512;            // feature dim
constexpr int kH  = 8;              // heads
constexpr int kC  = 64;             // channels per head
constexpr int kE2 = kE + kN;        // edges incl. self loops = 200000
constexpr int kEAR = kE + 16;       // edge-feature rows incl. 16 mean rows
constexpr int kNC = 8;              // node classes
constexpr int kEC = 6;              // edge classes

typedef __attribute__((ext_vector_type(16))) __bf16 v16bf;
typedef __attribute__((ext_vector_type(8)))  float  v8f;

union FragAB { v16bf v; uint4 q[2]; };

__device__ inline unsigned short f2bf(float f) {
  union { float f; unsigned u; } c; c.f = f;
  unsigned u = c.u;
  unsigned r = u + 0x7FFFu + ((u >> 16) & 1u);
  return (unsigned short)(r >> 16);
}
__device__ inline v8f v8f_zero() {
  v8f r;
#pragma unroll
  for (int i = 0; i < 8; ++i) r[i] = 0.f;
  return r;
}

// Issue one per-lane 16B async copy global->LDS (tracked by ASYNCcnt).
__device__ inline void async_g2l_b128(unsigned lds_off, const void* gaddr) {
  asm volatile("global_load_async_to_lds_b128 %0, %1, off"
               :: "v"(lds_off), "v"((unsigned long long)(size_t)gaddr)
               : "memory");
}

// ---------------------------------------------------------------------------
// fp32 -> bf16 conversion (grid-stride)
__global__ __launch_bounds__(256)
void cvt_bf16_kernel(const float* __restrict__ in, unsigned short* __restrict__ out, size_t n) {
  size_t i = (size_t)blockIdx.x * blockDim.x + threadIdx.x;
  size_t stride = (size_t)gridDim.x * blockDim.x;
  for (; i < n; i += stride) out[i] = f2bf(in[i]);
}

// Column mean of edge_features -> fill 16 mean rows (bf16) at rows kE..kE+15
__global__ __launch_bounds__(256)
void col_mean_fill_kernel(const float* __restrict__ ef, unsigned short* __restrict__ ea_bf) {
  int c = blockIdx.x;                       // 0..511
  float s = 0.f;
  for (int r = threadIdx.x; r < kE; r += 256) s += ef[(size_t)r * kD + c];
  __shared__ float sm[256];
  sm[threadIdx.x] = s; __syncthreads();
  for (int o = 128; o > 0; o >>= 1) {
    if (threadIdx.x < o) sm[threadIdx.x] += sm[threadIdx.x + o];
    __syncthreads();
  }
  if (threadIdx.x == 0) {
    unsigned short m = f2bf(sm[0] * (1.f / (float)kE));
    for (int r = 0; r < 16; ++r) ea_bf[(size_t)(kE + r) * kD + c] = m;
  }
}

// Pack row-major fp32 weight W[K x Nw] into bf16 WMMA-B fragment order.
// Tile (kt,nt) is 32(K)x16(N); lane l holds col n=l&15, K-half l>>4,
// element j = K value (l>>4)*16+j within the tile. 16 bf16 contiguous per lane.
__global__ __launch_bounds__(256)
void pack_w_kernel(const float* __restrict__ W, unsigned short* __restrict__ Bp, int K, int Nw) {
  size_t i = (size_t)blockIdx.x * 256 + threadIdx.x;
  size_t total = (size_t)K * Nw;
  if (i >= total) return;
  int NT = Nw >> 4;
  size_t t = i >> 9;                 // tile id
  int rem = (int)(i & 511);
  int l = rem >> 4, j = rem & 15;
  int kt = (int)(t / NT), nt = (int)(t % NT);
  int r = kt * 32 + (l >> 4) * 16 + j;
  int c = nt * 16 + (l & 15);
  Bp[i] = f2bf(W[(size_t)r * Nw + c]);
}

// ---------------------------------------------------------------------------
// Tiled bf16 GEMM: C[M x Nw] = A[M x K] * B[K x Nw] via v_wmma_f32_16x16x32_bf16.
// Block 256 (8 waves): 128x128 block tile, wave tile 32x64 (2x4 subtiles).
// B tiles (8 KB per K-step, shared by all waves) are staged into LDS with
// double-buffered GLOBAL_LOAD_ASYNC_TO_LDS_B128 (ASYNCcnt pipelining).
__global__ __launch_bounds__(256)
void gemm_bf16_kernel(const unsigned short* __restrict__ A,
                      const unsigned short* __restrict__ Bp,
                      float* __restrict__ C, int M, int K, int Nw) {
  __shared__ alignas(16) unsigned short Bs[2][8 * 512];   // 2 x 8KB

  const int tid  = threadIdx.x;
  const int lane = tid & 31;
  const int wave = tid >> 5;
  const int waveM = wave & 3, waveN = wave >> 2;
  const int rowBase = blockIdx.x * 128 + waveM * 32;
  const int colBase = blockIdx.y * 128 + waveN * 64;
  const int NT = Nw >> 4;
  const int half = lane >> 4;
  const int l15 = lane & 15;
  const int tileBase = blockIdx.y * 8;      // first B column-tile of this block

  unsigned ldsBase0 = (unsigned)(size_t)&Bs[0][0];
  unsigned ldsBase1 = (unsigned)(size_t)&Bs[1][0];

  // Stage the 8 contiguous packed-B tiles of K-step kt into Bs[buf]:
  // 8192 B = 512 x b128 chunks, 2 chunks per thread.
  auto stage = [&](int kt, unsigned ldsBase) {
    const unsigned short* src = Bp + ((size_t)kt * NT + tileBase) * 512;
#pragma unroll
    for (int i = 0; i < 2; ++i) {
      int chunk = tid + i * 256;
      async_g2l_b128(ldsBase + chunk * 16, src + chunk * 8);
    }
  };

  v8f acc[2][4];
#pragma unroll
  for (int m = 0; m < 2; ++m)
#pragma unroll
    for (int t = 0; t < 4; ++t) acc[m][t] = v8f_zero();

  const unsigned short* aptr[2];
#pragma unroll
  for (int m = 0; m < 2; ++m) {
    int r = rowBase + m * 16 + l15;
    if (r >= M) r = M - 1;                       // clamp; store is guarded
    aptr[m] = A + (size_t)r * K + half * 16;
  }

  const int ktiles = K >> 5;
  stage(0, ldsBase0);

  for (int kt = 0; kt < ktiles; ++kt) {
    const int buf = kt & 1;
    if (kt + 1 < ktiles) {
      stage(kt + 1, buf ? ldsBase0 : ldsBase1);
      asm volatile("s_wait_asynccnt 0x2" ::: "memory");   // stage(kt) done
    } else {
      asm volatile("s_wait_asynccnt 0x0" ::: "memory");
    }
    __syncthreads();    // stage(kt) visible to all waves

    FragAB a[2], b[4];
#pragma unroll
    for (int m = 0; m < 2; ++m) {
      const unsigned short* q = aptr[m] + kt * 32;
      a[m].q[0] = *(const uint4*)q;
      a[m].q[1] = *(const uint4*)(q + 8);
      if (kt + 1 < ktiles) __builtin_prefetch(q + 32, 0, 3);   // near prefetch
    }
#pragma unroll
    for (int t = 0; t < 4; ++t) {
      const unsigned short* q = &Bs[buf][((size_t)(waveN * 4 + t) * 32 + lane) * 16];
      b[t].q[0] = *(const uint4*)q;
      b[t].q[1] = *(const uint4*)(q + 8);
    }
#pragma unroll
    for (int m = 0; m < 2; ++m)
#pragma unroll
      for (int t = 0; t < 4; ++t)
        acc[m][t] = __builtin_amdgcn_wmma_f32_16x16x32_bf16(
            false, a[m].v, false, b[t].v, (short)0, acc[m][t], false, false);

    __syncthreads();    // all reads of Bs[buf] done before it is overwritten
  }

#pragma unroll
  for (int m = 0; m < 2; ++m)
#pragma unroll
    for (int t = 0; t < 4; ++t)
#pragma unroll
      for (int v = 0; v < 8; ++v) {
        int r = rowBase + m * 16 + half * 8 + v;
        if (r < M) {
          int c = colBase + t * 16 + l15;
          C[(size_t)r * Nw + c] = acc[m][t][v];
        }
      }
}

// ---------------------------------------------------------------------------
// init helpers
__global__ __launch_bounds__(256)
void fill_bias_kernel(float* __restrict__ out, const float* __restrict__ b) {
  size_t i = (size_t)blockIdx.x * 256 + threadIdx.x;
  if (i >= (size_t)kN * kD) return;
  out[i] = b[i & (kD - 1)];
}
__global__ __launch_bounds__(256)
void fill_u32_kernel(unsigned* __restrict__ p, unsigned v, size_t n) {
  size_t i = (size_t)blockIdx.x * 256 + threadIdx.x;
  if (i < n) p[i] = v;
}

// ---------------------------------------------------------------------------
// Pass 1: per-edge attention logits + segment max (one wave per edge).
__global__ __launch_bounds__(256)
void score_kernel(const float* __restrict__ xl, const float* __restrict__ xr,
                  const float* __restrict__ em, const float* __restrict__ att,
                  const int* __restrict__ src0, const int* __restrict__ dst0,
                  float* __restrict__ score, unsigned* __restrict__ smax) {
  int gw = (int)(((size_t)blockIdx.x * blockDim.x + threadIdx.x) >> 5);
  if (gw >= kE2) return;
  int lane = threadIdx.x & 31;
  int e = gw, s, d;
  const float* erow;
  if (e < kE) { s = src0[e]; d = dst0[e]; erow = em + (size_t)e * kD; }
  else        { s = d = e - kE;           erow = em + (size_t)kE * kD; }
  int c0 = lane * 16;
  int h = lane >> 2;
  const float* pl = xl + (size_t)s * kD + c0;
  const float* pr = xr + (size_t)d * kD + c0;
  const float* pe = erow + c0;
  const float* pa = att + h * kC + (lane & 3) * 16;
  float acc = 0.f;
#pragma unroll
  for (int j = 0; j < 16; ++j) {
    float m = pl[j] + pr[j] + pe[j];
    m = m > 0.f ? m : 0.2f * m;          // leaky_relu(0.2)
    acc += pa[j] * m;
  }
  acc += __shfl_xor(acc, 1, 32);
  acc += __shfl_xor(acc, 2, 32);
  if ((lane & 3) == 0) {
    score[(size_t)e * kH + h] = acc;
    unsigned u = __float_as_uint(acc);
    unsigned enc = (u & 0x80000000u) ? ~u : (u | 0x80000000u);
    atomicMax(smax + (size_t)d * kH + h, enc);
  }
}

// Pass 2: exp(score - max) + segment sum
__global__ __launch_bounds__(256)
void softmax_num_kernel(const int* __restrict__ dst0, float* __restrict__ score,
                        const unsigned* __restrict__ smax, float* __restrict__ denom) {
  size_t i = (size_t)blockIdx.x * 256 + threadIdx.x;
  if (i >= (size_t)kE2 * kH) return;
  int e = (int)(i >> 3), h = (int)(i & 7);
  int d = (e < kE) ? dst0[e] : e - kE;
  unsigned u = smax[(size_t)d * kH + h];
  unsigned bits = (u & 0x80000000u) ? (u & 0x7FFFFFFFu) : ~u;
  float mx = __uint_as_float(bits);
  float ex = expf(score[i] - mx);
  score[i] = ex;
  atomicAdd(denom + (size_t)d * kH + h, ex);
}

// Pass 3: hacc[dst] += alpha * xl[src]  (one wave per edge)
__global__ __launch_bounds__(256)
void aggregate_kernel(const int* __restrict__ src0, const int* __restrict__ dst0,
                      const float* __restrict__ xl, const float* __restrict__ ex,
                      const float* __restrict__ denom, float* __restrict__ hacc) {
  int gw = (int)(((size_t)blockIdx.x * blockDim.x + threadIdx.x) >> 5);
  if (gw >= kE2) return;
  int lane = threadIdx.x & 31;
  int e = gw, s, d;
  if (e < kE) { s = src0[e]; d = dst0[e]; } else s = d = e - kE;
  int h = lane >> 2;
  float a = ex[(size_t)e * kH + h] / (denom[(size_t)d * kH + h] + 1e-16f);
  int c0 = lane * 16;
  const float* ps = xl + (size_t)s * kD + c0;
  float* po = hacc + (size_t)d * kD + c0;
#pragma unroll
  for (int j = 0; j < 16; ++j) atomicAdd(po + j, a * ps[j]);
}

// ---------------------------------------------------------------------------
// LayerNorm + ReLU (+ optional residual), writes bf16 and optional fp32.
__global__ __launch_bounds__(256)
void ln_relu_kernel(const float* __restrict__ hin, const float* __restrict__ g,
                    const float* __restrict__ b, const float* __restrict__ res,
                    unsigned short* __restrict__ out_bf, float* __restrict__ out_f) {
  int gw = (int)(((size_t)blockIdx.x * blockDim.x + threadIdx.x) >> 5);
  if (gw >= kN) return;
  int lane = threadIdx.x & 31;
  const float* row = hin + (size_t)gw * kD;
  int d0 = lane * 16;
  float v[16], s = 0.f, s2 = 0.f;
#pragma unroll
  for (int j = 0; j < 16; ++j) { v[j] = row[d0 + j]; s += v[j]; s2 += v[j] * v[j]; }
#pragma unroll
  for (int o = 16; o > 0; o >>= 1) { s += __shfl_xor(s, o, 32); s2 += __shfl_xor(s2, o, 32); }
  float mu = s * (1.f / kD);
  float var = s2 * (1.f / kD) - mu * mu;
  float rstd = rsqrtf(var + 1e-5f);
#pragma unroll
  for (int j = 0; j < 16; ++j) {
    int d = d0 + j;
    float o = (v[j] - mu) * rstd * g[d] + b[d];
    o = o > 0.f ? o : 0.f;
    if (res) o += res[(size_t)gw * kD + d];
    if (out_f) out_f[(size_t)gw * kD + d] = o;
    out_bf[(size_t)gw * kD + d] = f2bf(o);
  }
}

// ---------------------------------------------------------------------------
// Node head: softmax(h @ node_W + node_b), one wave per node.
__global__ __launch_bounds__(256)
void node_head_kernel(const float* __restrict__ h, const float* __restrict__ W,
                      const float* __restrict__ b, float* __restrict__ out) {
  int gw = (int)(((size_t)blockIdx.x * blockDim.x + threadIdx.x) >> 5);
  if (gw >= kN) return;
  int lane = threadIdx.x & 31;
  const float* row = h + (size_t)gw * kD;
  float acc[kNC];
#pragma unroll
  for (int c = 0; c < kNC; ++c) acc[c] = 0.f;
  for (int d = lane; d < kD; d += 32) {
    float hv = row[d];
    const float* wr = W + (size_t)d * kNC;
#pragma unroll
    for (int c = 0; c < kNC; ++c) acc[c] += hv * wr[c];
  }
#pragma unroll
  for (int c = 0; c < kNC; ++c)
#pragma unroll
    for (int o = 16; o > 0; o >>= 1) acc[c] += __shfl_xor(acc[c], o, 32);
  if (lane == 0) {
    float mx = -3.0e38f;
#pragma unroll
    for (int c = 0; c < kNC; ++c) { acc[c] += b[c]; mx = fmaxf(mx, acc[c]); }
    float se = 0.f;
#pragma unroll
    for (int c = 0; c < kNC; ++c) { acc[c] = expf(acc[c] - mx); se += acc[c]; }
    float inv = 1.f / se;
#pragma unroll
    for (int c = 0; c < kNC; ++c) out[(size_t)gw * kNC + c] = acc[c] * inv;
  }
}

// Edge class head: softmax(h[src] @ ecls_W + ecls_b), one wave per edge.
__global__ __launch_bounds__(256)
void edge_cls_kernel(const float* __restrict__ h, const int* __restrict__ src0,
                     const float* __restrict__ W, const float* __restrict__ b,
                     float* __restrict__ out) {
  int gw = (int)(((size_t)blockIdx.x * blockDim.x + threadIdx.x) >> 5);
  if (gw >= kE) return;
  int lane = threadIdx.x & 31;
  const float* row = h + (size_t)src0[gw] * kD;
  float acc[kEC];
#pragma unroll
  for (int c = 0; c < kEC; ++c) acc[c] = 0.f;
  for (int d = lane; d < kD; d += 32) {
    float hv = row[d];
    const float* wr = W + (size_t)d * kEC;
#pragma unroll
    for (int c = 0; c < kEC; ++c) acc[c] += hv * wr[c];
  }
#pragma unroll
  for (int c = 0; c < kEC; ++c)
#pragma unroll
    for (int o = 16; o > 0; o >>= 1) acc[c] += __shfl_xor(acc[c], o, 32);
  if (lane == 0) {
    float mx = -3.0e38f;
#pragma unroll
    for (int c = 0; c < kEC; ++c) { acc[c] += b[c]; mx = fmaxf(mx, acc[c]); }
    float se = 0.f;
#pragma unroll
    for (int c = 0; c < kEC; ++c) { acc[c] = expf(acc[c] - mx); se += acc[c]; }
    float inv = 1.f / se;
#pragma unroll
    for (int c = 0; c < kEC; ++c) out[(size_t)gw * kEC + c] = acc[c] * inv;
  }
}

// ---------------------------------------------------------------------------
// Fused edge-existence MLP: sigmoid(relu([h[src],h[dst]] @ W1 + b1) . W2 + b2)
// One block = 16 edges. 8 waves each own a 16x64 slice of the 512 hidden units.
// A is gathered from h_bf16 per edge; K 0..511 -> src row, 512..1023 -> dst row.
__global__ __launch_bounds__(256)
void edge_mlp_kernel(const unsigned short* __restrict__ hb,
                     const int* __restrict__ src0, const int* __restrict__ dst0,
                     const unsigned short* __restrict__ W1p,
                     const float* __restrict__ b1, const float* __restrict__ W2,
                     const float* __restrict__ b2, float* __restrict__ out) {
  const int lane = threadIdx.x & 31;
  const int wave = threadIdx.x >> 5;
  const int half = lane >> 4;
  const int l15 = lane & 15;

  int e = blockIdx.x * 16 + l15;
  const unsigned short* rs = hb + (size_t)src0[e] * kD;
  const unsigned short* rt = hb + (size_t)dst0[e] * kD;

  v8f acc[4];
#pragma unroll
  for (int t = 0; t < 4; ++t) acc[t] = v8f_zero();

  for (int kb = 0; kb < 2 * kD; kb += 32) {
    const unsigned short* base = (kb < kD) ? (rs + kb) : (rt + (kb - kD));
    FragAB a, b[4];
    const unsigned short* q = base + half * 16;
    a.q[0] = *(const uint4*)q;
    a.q[1] = *(const uint4*)(q + 8);
    int kt = kb >> 5;
#pragma unroll
    for (int t = 0; t < 4; ++t) {
      const unsigned short* bq = W1p + (((size_t)kt * 32 + wave * 4 + t) * 32 + lane) * 16;
      b[t].q[0] = *(const uint4*)bq;
      b[t].q[1] = *(const uint4*)(bq + 8);
    }
#pragma unroll
    for (int t = 0; t < 4; ++t)
      acc[t] = __builtin_amdgcn_wmma_f32_16x16x32_bf16(
          false, a.v, false, b[t].v, (short)0, acc[t], false, false);
  }

  // bias + relu + dot with W2, reduce per row
  float rsum[8];
#pragma unroll
  for (int v = 0; v < 8; ++v) rsum[v] = 0.f;
#pragma unroll
  for (int t = 0; t < 4; ++t) {
    int c = wave * 64 + t * 16 + l15;
    float b1v = b1[c], w2v = W2[c];
#pragma unroll
    for (int v = 0; v < 8; ++v) {
      float z = acc[t][v] + b1v;
      z = z > 0.f ? z : 0.f;
      rsum[v] += z * w2v;
    }
  }
#pragma unroll
  for (int v = 0; v < 8; ++v)
#pragma unroll
    for (int o = 1; o < 16; o <<= 1) rsum[v] += __shfl_xor(rsum[v], o, 32);

  __shared__ float red[8][16];
  if (l15 == 0) {
#pragma unroll
    for (int v = 0; v < 8; ++v) red[wave][v + half * 8] = rsum[v];
  }
  __syncthreads();
  if (threadIdx.x < 16) {
    float s = b2[0];
#pragma unroll
    for (int w = 0; w < 8; ++w) s += red[w][threadIdx.x];
    out[blockIdx.x * 16 + threadIdx.x] = 1.f / (1.f + expf(-s));
  }
}

// ---------------------------------------------------------------------------
extern "C" void kernel_launch(void* const* d_in, const int* in_sizes, int n_in,
                              void* d_out, int out_size, void* d_ws, size_t ws_size,
                              hipStream_t stream) {
  (void)in_sizes; (void)n_in; (void)out_size; (void)ws_size;
  const float* x     = (const float*)d_in[0];
  const float* ef    = (const float*)d_in[1];
  const int*   eidx  = (const int*)d_in[2];
  const float* g1Wl  = (const float*)d_in[3];
  const float* g1Wr  = (const float*)d_in[4];
  const float* g1We  = (const float*)d_in[5];
  const float* g1att = (const float*)d_in[6];
  const float* g1b   = (const float*)d_in[7];
  const float* ln1g  = (const float*)d_in[8];
  const float* ln1b  = (const float*)d_in[9];
  const float* g2Wl  = (const float*)d_in[10];
  const float* g2Wr  = (const float*)d_in[11];
  const float* g2We  = (const float*)d_in[12];
  const float* g2att = (const float*)d_in[13];
  const float* g2b   = (const float*)d_in[14];
  const float* ln2g  = (const float*)d_in[15];
  const float* ln2b  = (const float*)d_in[16];
  const float* nodeW = (const float*)d_in[17];
  const float* nodeb = (const float*)d_in[18];
  const float* eclsW = (const float*)d_in[19];
  const float* eclsb = (const float*)d_in[20];
  const float* mlpW1 = (const float*)d_in[21];
  const float* mlpb1 = (const float*)d_in[22];
  const float* mlpW2 = (const float*)d_in[23];
  const float* mlpb2 = (const float*)d_in[24];

  const int* src0 = eidx;
  const int* dst0 = eidx + kE;

  // --- workspace carve-up ---
  uint8_t* base = (uint8_t*)d_ws;
  size_t off = 0;
  auto alloc = [&](size_t bytes) -> void* {
    void* r = base + off;
    off = (off + bytes + 255) & ~(size_t)255;
    return r;
  };
  unsigned short* x_bf  = (unsigned short*)alloc((size_t)kN * kD * 2);
  unsigned short* ea_bf = (unsigned short*)alloc((size_t)kEAR * kD * 2);
  unsigned short* h_bf  = (unsigned short*)alloc((size_t)kN * kD * 2);
  unsigned short* wpk[6];
  for (int i = 0; i < 6; ++i) wpk[i] = (unsigned short*)alloc((size_t)kD * kD * 2);
  unsigned short* w1pk = (unsigned short*)alloc((size_t)(2 * kD) * kD * 2);
  float* xl    = (float*)alloc((size_t)kN * kD * 4);
  float* xr    = (float*)alloc((size_t)kN * kD * 4);   // reused as final h (fp32)
  float* em    = (float*)alloc((size_t)kEAR * kD * 4);
  float* hacc  = (float*)alloc((size_t)kN * kD * 4);
  float* ex    = (float*)alloc((size_t)kE2 * kH * 4);
  unsigned* smax = (unsigned*)alloc((size_t)kN * kH * 4);
  float* denom = (float*)alloc((size_t)kN * kH * 4);

  // --- conversions / packing ---
  {
    size_t nx = (size_t)kN * kD;
    cvt_bf16_kernel<<<(unsigned)((nx + 255) / 256), 256, 0, stream>>>(x, x_bf, nx);
    size_t ne = (size_t)kE * kD;
    cvt_bf16_kernel<<<(unsigned)((ne + 255) / 256), 256, 0, stream>>>(ef, ea_bf, ne);
    col_mean_fill_kernel<<<kD, 256, 0, stream>>>(ef, ea_bf);
    const float* Ws[6] = {g1Wl, g1Wr, g1We, g2Wl, g2Wr, g2We};
    for (int i = 0; i < 6; ++i)
      pack_w_kernel<<<(unsigned)(((size_t)kD * kD + 255) / 256), 256, 0, stream>>>(Ws[i], wpk[i], kD, kD);
    pack_w_kernel<<<(unsigned)(((size_t)2 * kD * kD + 255) / 256), 256, 0, stream>>>(mlpW1, w1pk, 2 * kD, kD);
  }

  auto gemm = [&](const unsigned short* A, const unsigned short* Bp, float* C, int M) {
    dim3 g((M + 127) / 128, kD / 128);
    gemm_bf16_kernel<<<g, 256, 0, stream>>>(A, Bp, C, M, kD, kD);
  };

  auto run_layer = [&](const unsigned short* in_bf, const unsigned short* wl,
                       const unsigned short* wr, const unsigned short* we,
                       const float* att, const float* bias) {
    gemm(in_bf, wl, xl, kN);
    gemm(in_bf, wr, xr, kN);
    gemm(ea_bf, we, em, kEAR);
    fill_bias_kernel<<<(unsigned)(((size_t)kN * kD + 255) / 256), 256, 0, stream>>>(hacc, bias);
    fill_u32_kernel<<<(unsigned)(((size_t)kN * kH + 255) / 256), 256, 0, stream>>>(smax, 0u, (size_t)kN * kH);
    fill_u32_kernel<<<(unsigned)(((size_t)kN * kH + 255) / 256), 256, 0, stream>>>((unsigned*)denom, 0u, (size_t)kN * kH);
    unsigned edgeBlocks = (unsigned)(((size_t)kE2 * 32 + 255) / 256);
    score_kernel<<<edgeBlocks, 256, 0, stream>>>(xl, xr, em, att, src0, dst0, ex, smax);
    softmax_num_kernel<<<(unsigned)(((size_t)kE2 * kH + 255) / 256), 256, 0, stream>>>(dst0, ex, smax, denom);
    aggregate_kernel<<<edgeBlocks, 256, 0, stream>>>(src0, dst0, xl, ex, denom, hacc);
  };

  unsigned nodeWaveBlocks = (unsigned)(((size_t)kN * 32 + 255) / 256);

  // Layer 1
  run_layer(x_bf, wpk[0], wpk[1], wpk[2], g1att, g1b);
  ln_relu_kernel<<<nodeWaveBlocks, 256, 0, stream>>>(hacc, ln1g, ln1b, nullptr, h_bf, nullptr);

  // Layer 2 (consumes h_bf from layer 1)
  run_layer(h_bf, wpk[3], wpk[4], wpk[5], g2att, g2b);
  float* hf = xr;   // reuse xr as final fp32 h
  ln_relu_kernel<<<nodeWaveBlocks, 256, 0, stream>>>(hacc, ln2g, ln2b, x, h_bf, hf);

  // Heads
  float* outp = (float*)d_out;
  float* out_node = outp;
  float* out_et   = outp + (size_t)kN * kNC;
  float* out_ex   = outp + (size_t)kN * kNC + (size_t)kE * kEC;
  node_head_kernel<<<nodeWaveBlocks, 256, 0, stream>>>(hf, nodeW, nodeb, out_node);
  edge_cls_kernel<<<(unsigned)(((size_t)kE * 32 + 255) / 256), 256, 0, stream>>>(hf, src0, eclsW, eclsb, out_et);
  edge_mlp_kernel<<<kE / 16, 256, 0, stream>>>(h_bf, src0, dst0, w1pk, mlpb1, mlpW2, mlpb2, out_ex);
}
